// CrossAttention_53102975648312
// MI455X (gfx1250) — matrix operand order repaired and checked by reading, not compiled, not run on hardware
//
#include <hip/hip_runtime.h>
#include <cstddef>

// ---------------------------------------------------------------------------
// Bidirectional cross attention for MI455X (gfx1250, wave32, WMMA).
// Pipeline: f32->f16 convert (+weight transpose) -> WMMA qkv GEMMs
//           (16x64 tile per wave; batched fragment loads so the scheduler
//           can overlap global_load_b128 clauses with the matrix pipe)
//           -> wave-per-Qtile flash attention (v_wmma_f32_16x16x32_f16,
//              ds_swizzle row reductions, LDS C->A reshuffle, s_wait_dscnt)
//           -> WMMA output projection with f32+bias epilogue.
// ---------------------------------------------------------------------------

typedef _Float16 half8  __attribute__((ext_vector_type(8)));
typedef _Float16 half16 __attribute__((ext_vector_type(16)));
typedef float    float8 __attribute__((ext_vector_type(8)));

#define D_DIM 256
#define TK 32
#define NT 4  // N-tiles (of 16) per wave in the GEMM kernels

static __device__ __forceinline__ half16 cat16(half8 lo, half8 hi) {
  return __builtin_shufflevector(lo, hi, 0, 1, 2, 3, 4, 5, 6, 7, 8, 9, 10, 11,
                                 12, 13, 14, 15);
}

static __device__ __forceinline__ float8 wmma16(half16 a, half16 b, float8 c) {
  // (neg_a, A, neg_b, B, c_mod, C, reuse_a, reuse_b)
  return __builtin_amdgcn_wmma_f32_16x16x32_f16(false, a, false, b, (short)0, c,
                                                false, false);
}

// Load one 16-wide B fragment (16 contiguous halves) as two b128 loads.
static __device__ __forceinline__ half16 ldB(const _Float16* p) {
  return cat16(*(const half8*)p, *(const half8*)(p + 8));
}
// Load one A fragment: 16 contiguous halves at +2*koffA.
static __device__ __forceinline__ half16 ldA(const _Float16* p, int koffA) {
  return cat16(*(const half8*)(p + 2 * koffA), *(const half8*)(p + 2 * koffA + 8));
}

// xor-shuffle within the wave via ds_swizzle (group-of-32: offset =
// xor_mask<<10 | and_mask 0x1f). Masks <= 8 never cross a 16-lane half.
template <int MASK>
static __device__ __forceinline__ float shflx(float v) {
  int i = __builtin_amdgcn_ds_swizzle(__builtin_bit_cast(int, v),
                                      (MASK << 10) | 0x1f);
  return __builtin_bit_cast(float, i);
}

// ---------------------------------------------------------------------------
__global__ __launch_bounds__(256) void cvt_f32_f16(const float* __restrict__ in,
                                                   _Float16* __restrict__ out,
                                                   int n) {
  int i = blockIdx.x * blockDim.x + threadIdx.x;
  if (i < n) out[i] = (_Float16)in[i];
}

// W is [K][N] row-major (in,out). WT becomes [N][K] row-major.
__global__ __launch_bounds__(256) void transpose_f32_f16(
    const float* __restrict__ W, _Float16* __restrict__ WT, int K, int N) {
  int i = blockIdx.x * blockDim.x + threadIdx.x;
  if (i < K * N) {
    int k = i / N, n = i - k * N;
    WT[(size_t)n * K + k] = (_Float16)W[i];
  }
}

// ---------------------------------------------------------------------------
// C[M,N] = A[M,K] @ BT[N,K]^T. One wave computes a 16 x (16*NT) tile; all
// fragments of a k-chunk are loaded into distinct values before the WMMAs so
// the scheduler can clause the loads and use partial loadcnt waits.
// transS==0: store C row-major f16. transS==S: store per-batch transposed
// ([B][N][S]) for the V matrices consumed by the PV WMMA.
__global__ __launch_bounds__(256) void gemm_wmma_f16(
    const _Float16* __restrict__ A, const _Float16* __restrict__ BT,
    _Float16* __restrict__ C, int M, int N, int K, int transS) {
  const int lane = threadIdx.x & 31;
  const int gwave = (int)((blockIdx.x * (unsigned)blockDim.x + threadIdx.x) >> 5);
  const int ntiles = N / (16 * NT);
  const int total = (M >> 4) * ntiles;
  if (gwave >= total) return;  // wave-uniform
  const int mt = gwave / ntiles, nt = gwave - mt * ntiles;
  const int hi = lane >> 4, l15 = lane & 15;
  const int koffA = hi << 3;   // A frag: 16 contiguous halves at 2*koffA
  const int koffB = hi << 4;   // B frag: 16 contiguous halves at koffB
  const _Float16* pa = A + (size_t)(mt * 16 + l15) * K;
  const _Float16* pb0 = BT + (size_t)(nt * 16 * NT + l15) * K + koffB;
  float8 acc[NT];
#pragma unroll
  for (int u = 0; u < NT; ++u) acc[u] = float8{};
  for (int k0 = 0; k0 < K; k0 += 32) {
    half16 a = ldA(pa + k0, koffA);
    half16 b[NT];
#pragma unroll
    for (int u = 0; u < NT; ++u) b[u] = ldB(pb0 + (size_t)(u * 16) * K + k0);
#pragma unroll
    for (int u = 0; u < NT; ++u) acc[u] = wmma16(a, b[u], acc[u]);
  }
  if (transS == 0) {
#pragma unroll
    for (int u = 0; u < NT; ++u) {
      int n = (nt * NT + u) * 16 + l15;
#pragma unroll
      for (int r = 0; r < 8; ++r)
        C[(size_t)(mt * 16 + r + 8 * hi) * N + n] = (_Float16)acc[u][r];
    }
  } else {
#pragma unroll
    for (int u = 0; u < NT; ++u) {
      int n = (nt * NT + u) * 16 + l15;
#pragma unroll
      for (int r = 0; r < 8; ++r) {
        int m = mt * 16 + r + 8 * hi;
        int bb = m / transS, s = m - bb * transS;
        C[(size_t)bb * N * transS + (size_t)n * transS + s] =
            (_Float16)acc[u][r];
      }
    }
  }
}

// out[M,N] (f32) = A[M,K] @ BT[N,K]^T + bias[N]; same 16x(16*NT) tiling.
__global__ __launch_bounds__(256) void gemm_wmma_bias_f32(
    const _Float16* __restrict__ A, const _Float16* __restrict__ BT,
    const float* __restrict__ bias, float* __restrict__ C, int M, int N,
    int K) {
  const int lane = threadIdx.x & 31;
  const int gwave = (int)((blockIdx.x * (unsigned)blockDim.x + threadIdx.x) >> 5);
  const int ntiles = N / (16 * NT);
  const int total = (M >> 4) * ntiles;
  if (gwave >= total) return;
  const int mt = gwave / ntiles, nt = gwave - mt * ntiles;
  const int hi = lane >> 4, l15 = lane & 15;
  const int koffA = hi << 3, koffB = hi << 4;
  const _Float16* pa = A + (size_t)(mt * 16 + l15) * K;
  const _Float16* pb0 = BT + (size_t)(nt * 16 * NT + l15) * K + koffB;
  float8 acc[NT];
#pragma unroll
  for (int u = 0; u < NT; ++u) acc[u] = float8{};
  for (int k0 = 0; k0 < K; k0 += 32) {
    half16 a = ldA(pa + k0, koffA);
    half16 b[NT];
#pragma unroll
    for (int u = 0; u < NT; ++u) b[u] = ldB(pb0 + (size_t)(u * 16) * K + k0);
#pragma unroll
    for (int u = 0; u < NT; ++u) acc[u] = wmma16(a, b[u], acc[u]);
  }
#pragma unroll
  for (int u = 0; u < NT; ++u) {
    int n = (nt * NT + u) * 16 + l15;
    float bv = bias[n];
#pragma unroll
    for (int r = 0; r < 8; ++r)
      C[(size_t)(mt * 16 + r + 8 * hi) * N + n] = acc[u][r] + bv;
  }
}

// ---------------------------------------------------------------------------
// Flash attention, one wave per 16 query rows, full D=256 resident.
// Q,K: [B][S][256] f16 row-major. VT: [B][256][S] f16. O: [B][S][256] f16.
__global__ __launch_bounds__(128) void flash_attn(
    const _Float16* __restrict__ Q, const _Float16* __restrict__ Km,
    const _Float16* __restrict__ VT, _Float16* __restrict__ O, int S, int B) {
  __shared__ alignas(16) _Float16 pstage[4][16 * TK];
  const int lane = threadIdx.x & 31;
  const int wib = threadIdx.x >> 5;
  const int gwave = blockIdx.x * 4 + wib;
  const int tilesPerBatch = S >> 4;
  const int b = gwave / tilesPerBatch;
  const int qt = gwave - b * tilesPerBatch;
  if (b >= B) return;  // wave-uniform
  const int hi = lane >> 4, l15 = lane & 15;
  const int koffA = hi << 3, koffB = hi << 4;

  const _Float16* Qb = Q + (size_t)b * S * D_DIM;
  const _Float16* Kb = Km + (size_t)b * S * D_DIM;
  const _Float16* Vb = VT + (size_t)b * D_DIM * S;
  _Float16* Ob = O + (size_t)b * S * D_DIM;

  // Resident Q A-fragments: 8 chunks of K=32 over d.
  half16 aQ[8];
  {
    const _Float16* pq = Qb + (size_t)(qt * 16 + l15) * D_DIM;
#pragma unroll
    for (int c = 0; c < 8; ++c) aQ[c] = ldA(pq + c * 32, koffA);
  }

  float8 acc[16];
#pragma unroll
  for (int t = 0; t < 16; ++t) acc[t] = float8{};
  float mrow[8], lrow[8];
#pragma unroll
  for (int r = 0; r < 8; ++r) { mrow[r] = -1e30f; lrow[r] = 0.f; }

  const float scale = 0.0625f;  // 1/sqrt(256)
  _Float16* pb = &pstage[wib][0];

  for (int j = 0; j < S / TK; ++j) {
    // ---- scores: 16 x 32 keys, two 16x16 C fragments ----
    float8 s0 = {}, s1 = {};
    const _Float16* pk0 = Kb + (size_t)(j * TK + l15) * D_DIM + koffB;
    const _Float16* pk1 = Kb + (size_t)(j * TK + 16 + l15) * D_DIM + koffB;
    // two d-chunks per group: 4 B fragments in flight, then 4 WMMAs
#pragma unroll
    for (int c = 0; c < 8; c += 2) {
      half16 b00 = ldB(pk0 + c * 32);
      half16 b01 = ldB(pk1 + c * 32);
      half16 b10 = ldB(pk0 + (c + 1) * 32);
      half16 b11 = ldB(pk1 + (c + 1) * 32);
      s0 = wmma16(aQ[c], b00, s0);
      s1 = wmma16(aQ[c], b01, s1);
      s0 = wmma16(aQ[c + 1], b10, s0);
      s1 = wmma16(aQ[c + 1], b11, s1);
    }
    // ---- online softmax over the 32-key tile ----
#pragma unroll
    for (int r = 0; r < 8; ++r) {
      float a0 = s0[r] * scale, a1 = s1[r] * scale;
      float mx = fmaxf(a0, a1);
      mx = fmaxf(mx, shflx<1>(mx));
      mx = fmaxf(mx, shflx<2>(mx));
      mx = fmaxf(mx, shflx<4>(mx));
      mx = fmaxf(mx, shflx<8>(mx));
      float mnew = fmaxf(mrow[r], mx);
      float alpha = __expf(mrow[r] - mnew);
      float p0 = __expf(a0 - mnew);
      float p1 = __expf(a1 - mnew);
      float rs = p0 + p1;
      rs += shflx<1>(rs);
      rs += shflx<2>(rs);
      rs += shflx<4>(rs);
      rs += shflx<8>(rs);
      lrow[r] = lrow[r] * alpha + rs;
      mrow[r] = mnew;
#pragma unroll
      for (int t = 0; t < 16; ++t) acc[t][r] *= alpha;
      // stage P tile (C layout -> row-major 16x32 in LDS)
      int m = r + 8 * hi;
      pb[m * TK + l15] = (_Float16)p0;
      pb[m * TK + 16 + l15] = (_Float16)p1;
    }
    asm volatile("s_wait_dscnt 0" ::: "memory");  // intra-wave LDS visibility
    // read P back as one 16x32 A fragment
    half16 aP = ldA(pb + l15 * TK, koffA);
    // ---- PV: accumulate into all 16 d-tiles, 4 V fragments in flight ----
#pragma unroll
    for (int t0 = 0; t0 < 16; t0 += 4) {
      half16 bv[4];
#pragma unroll
      for (int t = 0; t < 4; ++t)
        bv[t] = ldB(Vb + (size_t)((t0 + t) * 16 + l15) * S + j * TK + koffB);
#pragma unroll
      for (int t = 0; t < 4; ++t)
        acc[t0 + t] = wmma16(aP, bv[t], acc[t0 + t]);
    }
  }

  // epilogue: normalize and store f16
#pragma unroll
  for (int r = 0; r < 8; ++r) {
    float inv = 1.f / lrow[r];
    int m = qt * 16 + r + 8 * hi;
#pragma unroll
    for (int t = 0; t < 16; ++t)
      Ob[(size_t)m * D_DIM + t * 16 + l15] = (_Float16)(acc[t][r] * inv);
  }
}

// ---------------------------------------------------------------------------
extern "C" void kernel_launch(void* const* d_in, const int* in_sizes, int n_in,
                              void* d_out, int out_size, void* d_ws,
                              size_t ws_size, hipStream_t stream) {
  (void)in_sizes; (void)n_in; (void)out_size; (void)ws_size;
  const int Bb = 8, S = 2048, Dd = 256, IN2 = 768;
  const int M = Bb * S;  // 16384

  const float* x1 = (const float*)d_in[0];
  const float* x2 = (const float*)d_in[1];
  const float* Wq1 = (const float*)d_in[2];
  const float* Wk1 = (const float*)d_in[3];
  const float* Wv1 = (const float*)d_in[4];
  const float* Wq2 = (const float*)d_in[5];
  const float* Wk2 = (const float*)d_in[6];
  const float* Wv2 = (const float*)d_in[7];
  const float* Wo1 = (const float*)d_in[8];
  const float* bo1 = (const float*)d_in[9];
  const float* Wo2 = (const float*)d_in[10];
  const float* bo2 = (const float*)d_in[11];
  float* out1 = (float*)d_out;                       // [16384][256]
  float* out2 = (float*)d_out + (size_t)M * Dd;      // [16384][768]

  size_t off = 0;
  char* wsb = (char*)d_ws;
  auto alloc_h = [&](size_t n) {
    _Float16* p = (_Float16*)(wsb + off);
    off += (n * sizeof(_Float16) + 255) & ~(size_t)255;
    return p;
  };
  _Float16* x1h = alloc_h((size_t)M * Dd);
  _Float16* x2h = alloc_h((size_t)M * IN2);
  _Float16* q1 = alloc_h((size_t)M * Dd);
  _Float16* k1 = alloc_h((size_t)M * Dd);
  _Float16* v1t = alloc_h((size_t)M * Dd);   // [B][256][S]
  _Float16* q2 = alloc_h((size_t)M * Dd);
  _Float16* k2 = alloc_h((size_t)M * Dd);
  _Float16* v2t = alloc_h((size_t)M * Dd);   // [B][256][S]
  _Float16* o1 = alloc_h((size_t)M * Dd);
  _Float16* o2 = alloc_h((size_t)M * Dd);
  _Float16* Wq1T = alloc_h((size_t)Dd * Dd);
  _Float16* Wk1T = alloc_h((size_t)Dd * Dd);
  _Float16* Wv1T = alloc_h((size_t)Dd * Dd);
  _Float16* Wq2T = alloc_h((size_t)Dd * IN2);
  _Float16* Wk2T = alloc_h((size_t)Dd * IN2);
  _Float16* Wv2T = alloc_h((size_t)Dd * IN2);
  _Float16* Wo1T = alloc_h((size_t)Dd * Dd);
  _Float16* Wo2T = alloc_h((size_t)IN2 * Dd);

  // 1) convert activations, transpose+convert weights
  cvt_f32_f16<<<(M * Dd + 255) / 256, 256, 0, stream>>>(x1, x1h, M * Dd);
  cvt_f32_f16<<<(M * IN2 + 255) / 256, 256, 0, stream>>>(x2, x2h, M * IN2);
  int w1n = Dd * Dd, w2n = IN2 * Dd;
  transpose_f32_f16<<<(w1n + 255) / 256, 256, 0, stream>>>(Wq1, Wq1T, Dd, Dd);
  transpose_f32_f16<<<(w1n + 255) / 256, 256, 0, stream>>>(Wk1, Wk1T, Dd, Dd);
  transpose_f32_f16<<<(w1n + 255) / 256, 256, 0, stream>>>(Wv1, Wv1T, Dd, Dd);
  transpose_f32_f16<<<(w2n + 255) / 256, 256, 0, stream>>>(Wq2, Wq2T, IN2, Dd);
  transpose_f32_f16<<<(w2n + 255) / 256, 256, 0, stream>>>(Wk2, Wk2T, IN2, Dd);
  transpose_f32_f16<<<(w2n + 255) / 256, 256, 0, stream>>>(Wv2, Wv2T, IN2, Dd);
  transpose_f32_f16<<<(w1n + 255) / 256, 256, 0, stream>>>(Wo1, Wo1T, Dd, Dd);
  transpose_f32_f16<<<(w2n + 255) / 256, 256, 0, stream>>>(Wo2, Wo2T, Dd, IN2);

  // 2) qkv projections: 16x64 tile per wave, 8 waves/block
  int t1 = (M / 16) * (Dd / (16 * NT));  // 4096 waves
  gemm_wmma_f16<<<t1 / 8, 256, 0, stream>>>(x1h, Wq1T, q1, M, Dd, Dd, 0);
  gemm_wmma_f16<<<t1 / 8, 256, 0, stream>>>(x1h, Wk1T, k1, M, Dd, Dd, 0);
  gemm_wmma_f16<<<t1 / 8, 256, 0, stream>>>(x1h, Wv1T, v1t, M, Dd, Dd, S);
  gemm_wmma_f16<<<t1 / 8, 256, 0, stream>>>(x2h, Wq2T, q2, M, Dd, IN2, 0);
  gemm_wmma_f16<<<t1 / 8, 256, 0, stream>>>(x2h, Wk2T, k2, M, Dd, IN2, 0);
  gemm_wmma_f16<<<t1 / 8, 256, 0, stream>>>(x2h, Wv2T, v2t, M, Dd, IN2, S);

  // 3) bidirectional flash attention (4 waves/block, 1 Q-tile/wave)
  int awaves = Bb * (S / 16);  // 1024
  flash_attn<<<awaves / 4, 128, 0, stream>>>(q1, k2, v2t, o1, S, Bb);
  flash_attn<<<awaves / 4, 128, 0, stream>>>(q2, k1, v1t, o2, S, Bb);

  // 4) output projections with bias, f32 into d_out
  gemm_wmma_bias_f32<<<t1 / 8, 256, 0, stream>>>(o1, Wo1T, bo1, out1, M, Dd, Dd);
  int t2 = (M / 16) * (IN2 / (16 * NT));  // 12288 waves
  gemm_wmma_bias_f32<<<t2 / 8, 256, 0, stream>>>(o2, Wo2T, bo2, out2, M, IN2, Dd);
}
